// QuantizedAntPolicy_3513283248696
// MI455X (gfx1250) — 8-bit matrix operand order repaired and checked by reading, compiled, not run on hardware
//
#include <hip/hip_runtime.h>
#include <stdint.h>

typedef int v8i __attribute__((ext_vector_type(8)));
typedef int v4i __attribute__((ext_vector_type(4)));
typedef int v2i __attribute__((ext_vector_type(2)));

#define AS1 __attribute__((address_space(1)))
#define AS3 __attribute__((address_space(3)))

__device__ __forceinline__ void cp_async_b128(void* lds, const void* g) {
#if __has_builtin(__builtin_amdgcn_global_load_async_to_lds_b128)
    __builtin_amdgcn_global_load_async_to_lds_b128(
        (AS1 v4i*)(void*)g, (AS3 v4i*)lds, 0, 0);
#else
    *(v4i*)lds = *(const v4i*)g;
#endif
}

__device__ __forceinline__ void async_wait0() {
#if __has_builtin(__builtin_amdgcn_s_wait_asynccnt)
    __builtin_amdgcn_s_wait_asynccnt(0);
#else
    asm volatile("s_wait_asynccnt 0" ::: "memory");
#endif
}

__global__ __launch_bounds__(32)
void init_slots_kernel(unsigned* __restrict__ slots) {
    if (threadIdx.x < 8) slots[threadIdx.x] = 0u;
}

__device__ __forceinline__ float slot_scale(const unsigned* __restrict__ slots, int idx) {
    float a = __uint_as_float(slots[idx]);
    return fmaxf(a * (1.0f / 127.0f), 1e-8f);
}

__global__ __launch_bounds__(256)
void absmax_kernel(const float4* __restrict__ x, long n4, unsigned* __restrict__ slot) {
    __shared__ unsigned red[256];
    unsigned m = 0u;
    long i = (long)blockIdx.x * 256 + threadIdx.x;
    const long stride = (long)gridDim.x * 256;
    for (; i < n4; i += stride) {
        float4 v = x[i];
        m = max(m, __float_as_uint(fabsf(v.x)));
        m = max(m, __float_as_uint(fabsf(v.y)));
        m = max(m, __float_as_uint(fabsf(v.z)));
        m = max(m, __float_as_uint(fabsf(v.w)));
    }
    red[threadIdx.x] = m;
    __syncthreads();
    for (int s = 128; s > 0; s >>= 1) {
        if ((int)threadIdx.x < s) red[threadIdx.x] = max(red[threadIdx.x], red[threadIdx.x + s]);
        __syncthreads();
    }
    if (threadIdx.x == 0) atomicMax(slot, red[0]);
}

__global__ __launch_bounds__(256)
void quant_kernel(const float4* __restrict__ x, long n4,
                  const unsigned* __restrict__ slots, int sidx,
                  int* __restrict__ q) {
    const float inv = 1.0f / slot_scale(slots, sidx);
    long i = (long)blockIdx.x * 256 + threadIdx.x;
    const long stride = (long)gridDim.x * 256;
    for (; i < n4; i += stride) {
        float4 v = x[i];
        int q0 = min(127, max(-128, (int)rintf(v.x * inv)));
        int q1 = min(127, max(-128, (int)rintf(v.y * inv)));
        int q2 = min(127, max(-128, (int)rintf(v.z * inv)));
        int q3 = min(127, max(-128, (int)rintf(v.w * inv)));
        q[i] = (q0 & 255) | ((q1 & 255) << 8) | ((q2 & 255) << 16) | (q3 << 24);
    }
}

#define ASTR 80
#define SLAB (128 * ASTR)

__global__ __launch_bounds__(256)
void gemm_i8_wmma_kernel(const int8_t* __restrict__ A,
                         const int8_t* __restrict__ W,
                         const float*  __restrict__ bias,
                         float*        __restrict__ C,
                         int M, int N, int K,
                         const unsigned* __restrict__ slots, int sa_idx, int sb_idx,
                         unsigned* __restrict__ out_absmax) {
    __shared__ __align__(16) char smA[2][SLAB];
    __shared__ __align__(16) char smB[2][SLAB];

    const int tid   = threadIdx.x;
    const int lane  = tid & 31;
    const int wave  = tid >> 5;
    const int wm    = wave >> 2;
    const int wn    = wave & 3;
    const int l16   = lane & 15;
    const int lhalf = lane >> 4;

    const int m0w = blockIdx.y * 128 + wm * 64;
    const int n0w = blockIdx.x * 128 + wn * 32;

    const int8_t* gA = A + (long)blockIdx.y * 128 * K;
    const int8_t* gB = W + (long)blockIdx.x * 128 * K;

    const int crow0 = tid >> 2;
    const int ccol  = (tid & 3) * 16;

    const v8i vzero = {0, 0, 0, 0, 0, 0, 0, 0};
    v8i acc[4][2];
#pragma unroll
    for (int i = 0; i < 4; ++i)
#pragma unroll
        for (int j = 0; j < 2; ++j) acc[i][j] = vzero;

    const int aOff = (wm * 64 + l16) * ASTR + lhalf * 16;
    const int bOff = (wn * 32 + l16) * ASTR + lhalf * 16;

    const int nk = K >> 6;

    {
        char* sA = smA[0];
        char* sB = smB[0];
#pragma unroll
        for (int it = 0; it < 2; ++it) {
            const int row = crow0 + it * 64;
            cp_async_b128(sA + row * ASTR + ccol, gA + (long)row * K + ccol);
            cp_async_b128(sB + row * ASTR + ccol, gB + (long)row * K + ccol);
        }
    }
    async_wait0();
    __syncthreads();

    for (int s = 0; s < nk; ++s) {
        if (s + 1 < nk) {
            const int k1 = (s + 1) << 6;
            char* sA = smA[(s + 1) & 1];
            char* sB = smB[(s + 1) & 1];
#pragma unroll
            for (int it = 0; it < 2; ++it) {
                const int row = crow0 + it * 64;
                cp_async_b128(sA + row * ASTR + ccol, gA + (long)row * K + k1 + ccol);
                cp_async_b128(sB + row * ASTR + ccol, gB + (long)row * K + k1 + ccol);
            }
        }

        const char* sA = smA[s & 1];
        const char* sB = smB[s & 1];

        v8i af[4];
#pragma unroll
        for (int i = 0; i < 4; ++i) {
            const char* p = sA + aOff + i * 16 * ASTR;
            v2i t0 = *(const v2i*)(p + 0);
            v2i t1 = *(const v2i*)(p + 8);
            v2i t2 = *(const v2i*)(p + 32);
            v2i t3 = *(const v2i*)(p + 40);
            v8i a;
            a[0] = t0[0]; a[1] = t0[1]; a[2] = t1[0]; a[3] = t1[1];
            a[4] = t2[0]; a[5] = t2[1]; a[6] = t3[0]; a[7] = t3[1];
            af[i] = a;
        }
        v8i bf[2];
#pragma unroll
        for (int j = 0; j < 2; ++j) {
            const char* p = sB + bOff + j * 16 * ASTR;
            v4i u0 = *(const v4i*)(p + 0);
            v4i u1 = *(const v4i*)(p + 32);
            v8i b;
            b[0] = u0[0]; b[1] = u0[1]; b[2] = u0[2]; b[3] = u0[3];
            b[4] = u1[0]; b[5] = u1[1]; b[6] = u1[2]; b[7] = u1[3];
            bf[j] = b;
        }
#pragma unroll
        for (int i = 0; i < 4; ++i)
#pragma unroll
            for (int j = 0; j < 2; ++j)
                acc[i][j] = __builtin_amdgcn_wmma_i32_16x16x64_iu8(
                     true, af[i],  true, bf[j],
                    acc[i][j],  false,  false);

        async_wait0();
        __syncthreads();
    }

    const float sc = slot_scale(slots, sa_idx) * slot_scale(slots, sb_idx);
    unsigned mloc = 0u;
#pragma unroll
    for (int j = 0; j < 2; ++j) {
        const int col = n0w + j * 16 + l16;
        const float bv = bias[col];
#pragma unroll
        for (int i = 0; i < 4; ++i) {
            const int rbase = m0w + i * 16 + lhalf * 8;
#pragma unroll
            for (int v = 0; v < 8; ++v) {
                float y = fmaf((float)acc[i][j][v], sc, bv);
                C[(long)(rbase + v) * N + col] = y;
                mloc = max(mloc, __float_as_uint(fabsf(y)));
            }
        }
    }

    if (out_absmax != nullptr) {
        __shared__ unsigned red[256];
        red[tid] = mloc;
        __syncthreads();
        for (int s = 128; s > 0; s >>= 1) {
            if (tid < s) red[tid] = max(red[tid], red[tid + s]);
            __syncthreads();
        }
        if (tid == 0) atomicMax(out_absmax, red[0]);
    }
}


static inline int rgrid(long n4) {
    long b = (n4 + 255) / 256;
    if (b > 4096) b = 4096;
    return (int)b;
}

extern "C" void kernel_launch(void* const* d_in, const int* in_sizes, int n_in,
                              void* d_out, int out_size, void* d_ws, size_t ws_size,
                              hipStream_t stream) {
    const float* x  = (const float*)d_in[0];
    const float* W1 = (const float*)d_in[1];
    const float* b1 = (const float*)d_in[2];
    const float* W2 = (const float*)d_in[3];
    const float* b2 = (const float*)d_in[4];
    const float* W3 = (const float*)d_in[5];
    const float* b3 = (const float*)d_in[6];

    const int H = in_sizes[2];
    const int O = in_sizes[6];
    const int D = in_sizes[1] / H;
    const int B = in_sizes[0] / D;

    char* ws = (char*)d_ws;
    size_t off = 0;
    auto carve = [&](size_t bytes) -> char* {
        char* p = ws + off;
        off = (off + bytes + 255) & ~(size_t)255;
        return p;
    };
    unsigned* slots = (unsigned*)carve(8 * sizeof(unsigned));
    int8_t* qW1 = (int8_t*)carve((size_t)H * D);
    int8_t* qW2 = (int8_t*)carve((size_t)H * H);
    int8_t* qW3 = (int8_t*)carve((size_t)O * H);
    int8_t* qA  = (int8_t*)carve((size_t)B * H);
    float*  Y   = (float*)carve((size_t)B * H * sizeof(float));
    (void)ws_size; (void)n_in; (void)out_size;

    const long nX  = (long)B * D, nW1 = (long)H * D, nW2 = (long)H * H, nW3 = (long)O * H;
    const long nY  = (long)B * H;

    init_slots_kernel<<<1, 32, 0, stream>>>(slots);

    absmax_kernel<<<rgrid(nX / 4),  256, 0, stream>>>((const float4*)x,  nX / 4,  slots + 0);
    absmax_kernel<<<rgrid(nW1 / 4), 256, 0, stream>>>((const float4*)W1, nW1 / 4, slots + 1);
    absmax_kernel<<<rgrid(nW2 / 4), 256, 0, stream>>>((const float4*)W2, nW2 / 4, slots + 2);
    absmax_kernel<<<rgrid(nW3 / 4), 256, 0, stream>>>((const float4*)W3, nW3 / 4, slots + 3);

    quant_kernel<<<rgrid(nX / 4),  256, 0, stream>>>((const float4*)x,  nX / 4,  slots, 0, (int*)qA);
    quant_kernel<<<rgrid(nW1 / 4), 256, 0, stream>>>((const float4*)W1, nW1 / 4, slots, 1, (int*)qW1);
    quant_kernel<<<rgrid(nW2 / 4), 256, 0, stream>>>((const float4*)W2, nW2 / 4, slots, 2, (int*)qW2);
    quant_kernel<<<rgrid(nW3 / 4), 256, 0, stream>>>((const float4*)W3, nW3 / 4, slots, 3, (int*)qW3);

    gemm_i8_wmma_kernel<<<dim3(H / 128, B / 128), 256, 0, stream>>>(
        qA, qW1, b1, Y, B, H, D, slots, 0, 1, slots + 4);
    quant_kernel<<<rgrid(nY / 4), 256, 0, stream>>>((const float4*)Y, nY / 4, slots, 4, (int*)qA);

    gemm_i8_wmma_kernel<<<dim3(H / 128, B / 128), 256, 0, stream>>>(
        qA, qW2, b2, Y, B, H, H, slots, 4, 2, slots + 5);
    quant_kernel<<<rgrid(nY / 4), 256, 0, stream>>>((const float4*)Y, nY / 4, slots, 5, (int*)qA);

    gemm_i8_wmma_kernel<<<dim3(O / 128, B / 128), 256, 0, stream>>>(
        qA, qW3, b3, (float*)d_out, B, O, H, slots, 5, 3, nullptr);
}
